// CausalSelfAttention_18064632447474
// MI455X (gfx1250) — matrix operand order repaired and checked by reading, not compiled, not run on hardware
//
#include <hip/hip_runtime.h>
#include <hip/hip_bf16.h>

typedef __attribute__((ext_vector_type(16))) __bf16 v16bf;
typedef __attribute__((ext_vector_type(8)))  __bf16 v8bf;
typedef __attribute__((ext_vector_type(8)))  float  v8f;

#define D_MODEL 1024
#define N_HEADS 16
#define HEAD_DIM 64
#define SEQ 2048
#define BATCH 4
#define MTOT (BATCH * SEQ) /* 8192 */

// ---------------- helpers ----------------

static __device__ __forceinline__ __bf16 f2bf(float f) {
  unsigned u = __builtin_bit_cast(unsigned, f);
  u += 0x7FFFu + ((u >> 16) & 1u);           // round-to-nearest-even
  unsigned short h = (unsigned short)(u >> 16);
  return __builtin_bit_cast(__bf16, h);
}

// A-operand fragment (16x32 bf16, M rows per lane):
// lane l<16 holds row M=l, K = {0..7, 16..23}; lanes 16..31 hold K = {8..15, 24..31}
static __device__ __forceinline__ v16bf ldfragA(const __bf16* p0, int ld, int lane) {
  const int l  = lane & 15;
  const int kh = (lane & 16) ? 8 : 0;
  const __bf16* p = p0 + (size_t)l * ld + kh;
  v8bf x0 = *(const v8bf*)(p);
  v8bf x1 = *(const v8bf*)(p + 16);
  v16bf r;
#pragma unroll
  for (int i = 0; i < 8; ++i) { r[i] = x0[i]; r[i + 8] = x1[i]; }
  return r;
}

// B-operand fragment (32x16 bf16, N columns per lane). Source rows = columns of B
// (i.e. rows of W for X*W^T, rows of K for Q*K^T, rows of V^T for P*V):
// lane l<16 holds column N=l, K = {0..7, 16..23}; lanes 16..31 hold K = {8..15, 24..31}
static __device__ __forceinline__ v16bf ldfragB(const __bf16* p0, int ld, int lane) {
  const int l  = lane & 15;
  const int kh = (lane & 16) ? 8 : 0;
  const __bf16* p = p0 + (size_t)l * ld + kh;
  v8bf x0 = *(const v8bf*)(p);
  v8bf x1 = *(const v8bf*)(p + 16);
  v16bf r;
#pragma unroll
  for (int i = 0; i < 8; ++i) { r[i] = x0[i]; r[i + 8] = x1[i]; }
  return r;
}

static __device__ __forceinline__ v8f wmma_bf16(v16bf a, v16bf b, v8f c) {
  return __builtin_amdgcn_wmma_f32_16x16x32_bf16(
      /*neg_a=*/false, a, /*neg_b=*/false, b,
      /*c_mod=*/(short)0, c, /*reuse_a=*/false, /*reuse_b=*/false);
}

// ---------------- f32 -> bf16 conversion ----------------

__global__ __launch_bounds__(256) void k_cvt(const float* __restrict__ in,
                                             __bf16* __restrict__ out, int n) {
  int i = blockIdx.x * 256 + threadIdx.x;
  if (i < n) out[i] = f2bf(in[i]);
}

// ---------------- GEMM mainloop: out = A(M,K) * W(N,K)^T ----------------
// One wave computes a 32x64 tile; block of 8 waves covers 128x128.

static __device__ __forceinline__ void gemm_tile(const __bf16* __restrict__ A,
                                                 const __bf16* __restrict__ W,
                                                 int K, int row0, int col0,
                                                 int lane, v8f (&acc)[2][4]) {
#pragma unroll 2
  for (int k0 = 0; k0 < K; k0 += 32) {
    v16bf a0 = ldfragA(A + (size_t)row0 * K + k0, K, lane);
    v16bf a1 = ldfragA(A + (size_t)(row0 + 16) * K + k0, K, lane);
    v16bf b0 = ldfragB(W + (size_t)(col0 +  0) * K + k0, K, lane);
    v16bf b1 = ldfragB(W + (size_t)(col0 + 16) * K + k0, K, lane);
    v16bf b2 = ldfragB(W + (size_t)(col0 + 32) * K + k0, K, lane);
    v16bf b3 = ldfragB(W + (size_t)(col0 + 48) * K + k0, K, lane);
    acc[0][0] = wmma_bf16(a0, b0, acc[0][0]);
    acc[0][1] = wmma_bf16(a0, b1, acc[0][1]);
    acc[0][2] = wmma_bf16(a0, b2, acc[0][2]);
    acc[0][3] = wmma_bf16(a0, b3, acc[0][3]);
    acc[1][0] = wmma_bf16(a1, b0, acc[1][0]);
    acc[1][1] = wmma_bf16(a1, b1, acc[1][1]);
    acc[1][2] = wmma_bf16(a1, b2, acc[1][2]);
    acc[1][3] = wmma_bf16(a1, b3, acc[1][3]);
  }
}

// ---------------- QKV projection: qkv = x @ W_qkv^T + b ----------------
// Scatters q,k as [B,h,T,64] bf16 and v transposed as [B,h,64,T] bf16.

__global__ __launch_bounds__(256) void k_gemm_qkv(const __bf16* __restrict__ xb,
                                                  const __bf16* __restrict__ wq,
                                                  const float* __restrict__ bias,
                                                  __bf16* __restrict__ qb,
                                                  __bf16* __restrict__ kb,
                                                  __bf16* __restrict__ vb) {
  const int lane = threadIdx.x & 31;
  const int wave = threadIdx.x >> 5;
  const int row0 = blockIdx.y * 128 + (wave & 3) * 32;
  const int col0 = blockIdx.x * 128 + (wave >> 2) * 64;
  v8f acc[2][4] = {};
  gemm_tile(xb, wq, D_MODEL, row0, col0, lane, acc);

  const int ln = lane & 15, hi = (lane >> 4) & 1;
#pragma unroll
  for (int mi = 0; mi < 2; ++mi)
#pragma unroll
    for (int ni = 0; ni < 4; ++ni)
#pragma unroll
      for (int i = 0; i < 8; ++i) {
        int row = row0 + mi * 16 + i + hi * 8;  // token index in [0, 8192)
        int col = col0 + ni * 16 + ln;          // output channel in [0, 3072)
        float v = acc[mi][ni][i] + bias[col];
        int which = col >> 10;                  // 0=q 1=k 2=v
        int c = col & 1023;
        int head = c >> 6, d = c & 63;
        int b = row >> 11, t = row & 2047;
        size_t bh = (size_t)b * N_HEADS + head;
        __bf16 bv = f2bf(v);
        if (which == 0)      qb[(bh * SEQ + t) * HEAD_DIM + d] = bv;
        else if (which == 1) kb[(bh * SEQ + t) * HEAD_DIM + d] = bv;
        else                 vb[(bh * HEAD_DIM + d) * SEQ + t] = bv;  // V^T
      }
}

// ---------------- Flash attention (causal, online softmax) ----------------
// 4 waves / block, each wave owns one 16-row query tile; keys in 32-wide tiles.

__global__ __launch_bounds__(128) void k_attn(const __bf16* __restrict__ qb,
                                              const __bf16* __restrict__ kb,
                                              const __bf16* __restrict__ vb,
                                              __bf16* __restrict__ yb) {
  __shared__ __attribute__((aligned(16))) __bf16 Pl[4][16 * 32];
  const int lane = threadIdx.x & 31;
  const int wave = threadIdx.x >> 5;
  const int ln = lane & 15, hi = (lane >> 4) & 1;
  const int bh = blockIdx.y;
  const int b = bh >> 4, head = bh & 15;
  const int qt0 = (blockIdx.x * 4 + wave) * 16;

  const __bf16* Q  = qb + (size_t)bh * SEQ * HEAD_DIM;
  const __bf16* Kp = kb + (size_t)bh * SEQ * HEAD_DIM;
  const __bf16* Vt = vb + (size_t)bh * HEAD_DIM * SEQ;  // [64][T]
  __bf16* Pw = &Pl[wave][0];

  // Q fragments are loop-invariant (hs = 64 -> two K=32 slabs)
  v16bf qa0 = ldfragA(Q + (size_t)qt0 * HEAD_DIM +  0, HEAD_DIM, lane);
  v16bf qa1 = ldfragA(Q + (size_t)qt0 * HEAD_DIM + 32, HEAD_DIM, lane);

  float rmax[8], rsum[8];
  v8f o[4] = {};
#pragma unroll
  for (int i = 0; i < 8; ++i) { rmax[i] = -1e30f; rsum[i] = 0.f; }

  const int kend = qt0 + 15;
#pragma unroll 1
  for (int kt = 0; kt <= kend; kt += 32) {
    if (kt + 32 <= kend)
      __builtin_prefetch(Kp + (size_t)(kt + 32) * HEAD_DIM, 0, 1);

    // S = Q K^T  (16x32 tile as two 16x16 D-frags)
    v8f s0 = {}, s1 = {};
    {
      v16bf bk0 = ldfragB(Kp + (size_t)(kt +  0) * HEAD_DIM +  0, HEAD_DIM, lane);
      v16bf bk1 = ldfragB(Kp + (size_t)(kt + 16) * HEAD_DIM +  0, HEAD_DIM, lane);
      s0 = wmma_bf16(qa0, bk0, s0);
      s1 = wmma_bf16(qa0, bk1, s1);
      v16bf ck0 = ldfragB(Kp + (size_t)(kt +  0) * HEAD_DIM + 32, HEAD_DIM, lane);
      v16bf ck1 = ldfragB(Kp + (size_t)(kt + 16) * HEAD_DIM + 32, HEAD_DIM, lane);
      s0 = wmma_bf16(qa1, ck0, s0);
      s1 = wmma_bf16(qa1, ck1, s1);
    }

    // scale + causal mask; per-row tile max
    float tmax[8];
#pragma unroll
    for (int i = 0; i < 8; ++i) {
      int qrow = qt0 + i + hi * 8;
      float v0 = s0[i] * 0.125f; if (kt + ln      > qrow) v0 = -1e30f;
      float v1 = s1[i] * 0.125f; if (kt + 16 + ln > qrow) v1 = -1e30f;
      s0[i] = v0; s1[i] = v1;
      tmax[i] = fmaxf(v0, v1);
    }
    // reduce across the 16-lane half (rows live per half in D-layout)
#pragma unroll
    for (int off = 8; off; off >>= 1)
#pragma unroll
      for (int i = 0; i < 8; ++i)
        tmax[i] = fmaxf(tmax[i], __shfl_xor(tmax[i], off, 32));

    float fs[8], ps[8];
#pragma unroll
    for (int i = 0; i < 8; ++i) {
      float nm = fmaxf(rmax[i], tmax[i]);
      fs[i] = __expf(rmax[i] - nm);
      rmax[i] = nm;
      float p0 = __expf(s0[i] - nm);
      float p1 = __expf(s1[i] - nm);
      ps[i] = p0 + p1;
      int m = i + hi * 8;
      Pw[m * 32 + ln]      = f2bf(p0);   // D-layout -> LDS (row-major 16x32)
      Pw[m * 32 + 16 + ln] = f2bf(p1);
    }
#pragma unroll
    for (int off = 8; off; off >>= 1)
#pragma unroll
      for (int i = 0; i < 8; ++i) ps[i] += __shfl_xor(ps[i], off, 32);
#pragma unroll
    for (int i = 0; i < 8; ++i) rsum[i] = rsum[i] * fs[i] + ps[i];
#pragma unroll
    for (int nf = 0; nf < 4; ++nf)
#pragma unroll
      for (int i = 0; i < 8; ++i) o[nf][i] *= fs[i];

    // O += P @ V : P reloaded as A-fragment (same-wave LDS ops are in-order)
    v16bf pf = ldfragA(Pw, 32, lane);
#pragma unroll
    for (int nf = 0; nf < 4; ++nf) {
      v16bf bv = ldfragB(Vt + (size_t)(nf * 16) * SEQ + kt, SEQ, lane);
      o[nf] = wmma_bf16(pf, bv, o[nf]);
    }
  }

  float inv[8];
#pragma unroll
  for (int i = 0; i < 8; ++i) inv[i] = 1.0f / rsum[i];
#pragma unroll
  for (int nf = 0; nf < 4; ++nf)
#pragma unroll
    for (int i = 0; i < 8; ++i) {
      int t = qt0 + i + hi * 8;
      int c = head * HEAD_DIM + nf * 16 + ln;
      yb[((size_t)b * SEQ + t) * D_MODEL + c] = f2bf(o[nf][i] * inv[i]);
    }
}

// ---------------- Output projection: out = y @ W_proj^T + b (f32 out) ----------------

__global__ __launch_bounds__(256) void k_gemm_proj(const __bf16* __restrict__ yb,
                                                   const __bf16* __restrict__ wp,
                                                   const float* __restrict__ bias,
                                                   float* __restrict__ out) {
  const int lane = threadIdx.x & 31;
  const int wave = threadIdx.x >> 5;
  const int row0 = blockIdx.y * 128 + (wave & 3) * 32;
  const int col0 = blockIdx.x * 128 + (wave >> 2) * 64;
  v8f acc[2][4] = {};
  gemm_tile(yb, wp, D_MODEL, row0, col0, lane, acc);

  const int ln = lane & 15, hi = (lane >> 4) & 1;
#pragma unroll
  for (int mi = 0; mi < 2; ++mi)
#pragma unroll
    for (int ni = 0; ni < 4; ++ni)
#pragma unroll
      for (int i = 0; i < 8; ++i) {
        int row = row0 + mi * 16 + i + hi * 8;
        int col = col0 + ni * 16 + ln;
        out[(size_t)row * D_MODEL + col] = acc[mi][ni][i] + bias[col];
      }
}

// ---------------- host launcher ----------------

extern "C" void kernel_launch(void* const* d_in, const int* in_sizes, int n_in,
                              void* d_out, int out_size, void* d_ws, size_t ws_size,
                              hipStream_t stream) {
  const float* x      = (const float*)d_in[0];
  const float* W_qkv  = (const float*)d_in[1];
  const float* b_qkv  = (const float*)d_in[2];
  const float* W_proj = (const float*)d_in[3];
  const float* b_proj = (const float*)d_in[4];
  float* out = (float*)d_out;

  char* ws = (char*)d_ws;
  size_t off = 0;
  auto carve = [&](size_t bytes) -> void* {
    void* p = ws + off;
    off += (bytes + 255) & ~(size_t)255;
    return p;
  };
  const size_t nx = (size_t)MTOT * D_MODEL;          // 8.39M
  const size_t nqkvw = (size_t)3 * D_MODEL * D_MODEL;
  const size_t npw = (size_t)D_MODEL * D_MODEL;
  const size_t nhd = (size_t)BATCH * N_HEADS * SEQ * HEAD_DIM;  // == nx

  __bf16* xb     = (__bf16*)carve(nx * 2);
  __bf16* wqkvb  = (__bf16*)carve(nqkvw * 2);
  __bf16* wprojb = (__bf16*)carve(npw * 2);
  __bf16* qb     = (__bf16*)carve(nhd * 2);
  __bf16* kb     = (__bf16*)carve(nhd * 2);
  __bf16* vb     = (__bf16*)carve(nhd * 2);
  __bf16* yb     = (__bf16*)carve(nx * 2);

  k_cvt<<<(int)((nx + 255) / 256), 256, 0, stream>>>(x, xb, (int)nx);
  k_cvt<<<(int)((nqkvw + 255) / 256), 256, 0, stream>>>(W_qkv, wqkvb, (int)nqkvw);
  k_cvt<<<(int)((npw + 255) / 256), 256, 0, stream>>>(W_proj, wprojb, (int)npw);

  // qkv = x @ W_qkv^T + b : M=8192, N=3072 -> grid (24, 64), 128x128 blocks
  k_gemm_qkv<<<dim3(3 * D_MODEL / 128, MTOT / 128), 256, 0, stream>>>(
      xb, wqkvb, b_qkv, qb, kb, vb);

  // attention: grid.x = (T/16)/4 query-tiles, grid.y = B*nh
  k_attn<<<dim3(SEQ / 64, BATCH * N_HEADS), 128, 0, stream>>>(qb, kb, vb, yb);

  // out = y @ W_proj^T + b : M=8192, N=1024 -> grid (8, 64)
  k_gemm_proj<<<dim3(D_MODEL / 128, MTOT / 128), 256, 0, stream>>>(
      yb, wprojb, b_proj, out);
}
